// GeometricNN_61881888801068
// MI455X (gfx1250) — matrix operand order repaired and checked by reading, not compiled, not run on hardware
//
#include <hip/hip_runtime.h>

#define DD 128
#define SLOPE 0.2f

typedef __bf16 bf16;
typedef __attribute__((ext_vector_type(16))) __bf16 v16bf;
typedef __attribute__((ext_vector_type(8)))  __bf16 v8bf;
typedef __attribute__((ext_vector_type(8)))  float  v8f;

#define WFRAG_ELEMS (4 * 8 * 32 * 16)          // per hi (or lo) plane
#define WFRAG_TOTAL (2 * WFRAG_ELEMS)          // hi + lo, per weight

// ---------------------------------------------------------------------------
// Float atomic max via sign-split int/uint ordering (init must be -inf).
// ---------------------------------------------------------------------------
__device__ inline void atomicMaxF(float* addr, float val) {
    float v = val + 0.0f;                 // canonicalize -0 -> +0
    if (v >= 0.0f) atomicMax((int*)addr, __float_as_int(v));
    else           atomicMin((unsigned int*)addr, __float_as_uint(v));
}

// ---------------------------------------------------------------------------
// Pack one 128x128 weight (row-major f32, W[k][n]) into WMMA B-fragment
// order: [kt][nt][lane][i] bf16, hi plane then lo plane.
// B fragment (16x16x32): lane l -> col = nt*16 + (l&15); element i -> K =
// kt*32 + ((l>>4)*8) + (i&7) + ((i>>3)*16).
// ---------------------------------------------------------------------------
__global__ void pack_w_frags(const float* __restrict__ W, bf16* __restrict__ out) {
    int idx = blockIdx.x * blockDim.x + threadIdx.x;
    if (idx >= WFRAG_ELEMS) return;
    int i    = idx & 15;
    int lane = (idx >> 4) & 31;
    int nt   = (idx >> 9) & 7;
    int kt   = idx >> 12;
    int col  = nt * 16 + (lane & 15);
    int k    = kt * 32 + ((lane >> 4) * 8) + (i & 7) + ((i >> 3) * 16);
    float wv = W[k * DD + col];
    bf16 h = (bf16)wv;
    out[idx]               = h;                       // hi plane
    out[WFRAG_ELEMS + idx] = (bf16)(wv - (float)h);   // lo plane
}

// ---------------------------------------------------------------------------
// Split f32 matrix into bf16 hi/lo planes (row-major).
// ---------------------------------------------------------------------------
__global__ void split_bf16(const float* __restrict__ X, bf16* __restrict__ hi,
                           bf16* __restrict__ lo, int total) {
    int i = blockIdx.x * blockDim.x + threadIdx.x;
    if (i >= total) return;
    float v = X[i];
    bf16 h = (bf16)v;
    hi[i] = h;
    lo[i] = (bf16)(v - (float)h);
}

// ---------------------------------------------------------------------------
// GEMM: C[M,128] = A[M,128] @ W[128,128] (+bias), split-bf16 x3 products.
// A pre-split into hi/lo bf16 planes; W pre-packed into fragment order.
// One wave: 16-row tile x 128 columns (8 accumulator tiles), K loop 4x32.
// Inner loop: pure vector loads + v_wmma_f32_16x16x32_bf16.
// ---------------------------------------------------------------------------
__global__ __launch_bounds__(256)
void gemm128_wmma(const bf16* __restrict__ Ahi, const bf16* __restrict__ Alo,
                  const bf16* __restrict__ Wfrag, const float* __restrict__ bias,
                  float* __restrict__ C, int M) {
    const int wave   = threadIdx.x >> 5;
    const int lane   = threadIdx.x & 31;
    const int rowT   = blockIdx.x * 8 + wave;
    if (rowT * 16 >= M) return;
    const int r0     = rowT * 16;
    const int rlane  = lane & 15;
    const int hiHalf = lane >> 4;
    const int kbA    = hiHalf * 8;         // A-fragment K base
    const bf16* ArowH = Ahi + (size_t)(r0 + rlane) * DD;
    const bf16* ArowL = Alo + (size_t)(r0 + rlane) * DD;
    const bf16* Whi   = Wfrag;
    const bf16* Wlo   = Wfrag + WFRAG_ELEMS;

    v8f acc[8] = {};

    for (int kt = 0; kt < 4; ++kt) {
        const int K0 = kt * 32;
        v8bf a0 = *(const v8bf*)(ArowH + K0 + kbA);
        v8bf a1 = *(const v8bf*)(ArowH + K0 + 16 + kbA);
        v8bf l0 = *(const v8bf*)(ArowL + K0 + kbA);
        v8bf l1 = *(const v8bf*)(ArowL + K0 + 16 + kbA);
        v16bf ahi = __builtin_shufflevector(a0, a1, 0,1,2,3,4,5,6,7,8,9,10,11,12,13,14,15);
        v16bf alo = __builtin_shufflevector(l0, l1, 0,1,2,3,4,5,6,7,8,9,10,11,12,13,14,15);
#pragma unroll
        for (int nt = 0; nt < 8; ++nt) {
            size_t off = (((size_t)kt * 8 + nt) * 32 + lane) * 16;
            v16bf bhi = *(const v16bf*)(Whi + off);
            v16bf blo = *(const v16bf*)(Wlo + off);
            acc[nt] = __builtin_amdgcn_wmma_f32_16x16x32_bf16(
                false, ahi, false, bhi, (short)0, acc[nt], false, false);
            acc[nt] = __builtin_amdgcn_wmma_f32_16x16x32_bf16(
                false, ahi, false, blo, (short)0, acc[nt], false, false);
            acc[nt] = __builtin_amdgcn_wmma_f32_16x16x32_bf16(
                false, alo, false, bhi, (short)0, acc[nt], false, false);
        }
    }
    // C/D layout: VGPR j -> M = j (lanes 0-15) / j+8 (lanes 16-31), N = lane&15
#pragma unroll
    for (int nt = 0; nt < 8; ++nt) {
        int col = nt * 16 + rlane;
        float badd = bias ? bias[col] : 0.0f;
#pragma unroll
        for (int j = 0; j < 8; ++j) {
            int row = r0 + j + hiHalf * 8;
            C[(size_t)row * DD + col] = acc[nt][j] + badd;
        }
    }
}

// ---------------------------------------------------------------------------
// Fill
// ---------------------------------------------------------------------------
__global__ void fill_f32(float* __restrict__ p, float v, int n) {
    int i = blockIdx.x * blockDim.x + threadIdx.x;
    if (i < n) p[i] = v;
}

// ---------------------------------------------------------------------------
// Edge pass 1: e = leakyrelu(xl[src]+xr[dst]) . att ; m[dst] = max(m[dst], e)
// ---------------------------------------------------------------------------
__global__ __launch_bounds__(256)
void edge_logits_max(const float* __restrict__ xl, const float* __restrict__ xr,
                     const float* __restrict__ att, const int* __restrict__ src,
                     const int* __restrict__ dst, float* __restrict__ e,
                     float* __restrict__ m, int E_) {
    int edge = (int)((blockIdx.x * (size_t)blockDim.x + threadIdx.x) >> 5);
    int lane = threadIdx.x & 31;
    if (edge >= E_) return;
    int s = src[edge], d = dst[edge];
    float4 u = ((const float4*)(xl + (size_t)s * DD))[lane];
    float4 v = ((const float4*)(xr + (size_t)d * DD))[lane];
    float4 a = ((const float4*)att)[lane];
    float t, sum = 0.0f;
    t = u.x + v.x; sum += (t > 0.0f ? t : SLOPE * t) * a.x;
    t = u.y + v.y; sum += (t > 0.0f ? t : SLOPE * t) * a.y;
    t = u.z + v.z; sum += (t > 0.0f ? t : SLOPE * t) * a.z;
    t = u.w + v.w; sum += (t > 0.0f ? t : SLOPE * t) * a.w;
    for (int off = 16; off > 0; off >>= 1) sum += __shfl_down(sum, off, 32);
    if (lane == 0) { e[edge] = sum; atomicMaxF(m + d, sum); }
}

// ---------------------------------------------------------------------------
// Edge pass 2: p = exp(e - m[dst]); s[dst]+=p; num[dst,:] += p * xl[src,:]
// ---------------------------------------------------------------------------
__global__ __launch_bounds__(256)
void edge_softmax_acc(const float* __restrict__ xl, const float* __restrict__ e,
                      const int* __restrict__ src, const int* __restrict__ dst,
                      const float* __restrict__ m, float* __restrict__ s,
                      float* __restrict__ num, int E_) {
    int edge = (int)((blockIdx.x * (size_t)blockDim.x + threadIdx.x) >> 5);
    int lane = threadIdx.x & 31;
    if (edge >= E_) return;
    int si = src[edge], d = dst[edge];
    float p = __expf(e[edge] - m[d]);
    if (lane == 0) atomicAdd(s + d, p);
    float4 u = ((const float4*)(xl + (size_t)si * DD))[lane];
    float* nd = num + (size_t)d * DD + lane * 4;
    atomicAdd(nd + 0, p * u.x);
    atomicAdd(nd + 1, p * u.y);
    atomicAdd(nd + 2, p * u.z);
    atomicAdd(nd + 3, p * u.w);
}

// ---------------------------------------------------------------------------
// Node combine: out = relu(num/(s+eps) + b + lin)
// ---------------------------------------------------------------------------
__global__ void node_combine_relu(const float* __restrict__ num, const float* __restrict__ s,
                                  const float* __restrict__ b, const float* __restrict__ lin,
                                  float* __restrict__ out, int total) {
    int i = blockIdx.x * blockDim.x + threadIdx.x;
    if (i >= total) return;
    int n = i >> 7, c = i & (DD - 1);
    float v = num[i] / (s[n] + 1e-16f) + b[c] + lin[i];
    out[i] = v > 0.0f ? v : 0.0f;
}

// ---------------------------------------------------------------------------
// Output head: three D->1 dots per node (wave-per-node, shuffle reduce)
// ---------------------------------------------------------------------------
__global__ __launch_bounds__(256)
void dot3_kernel(const float* __restrict__ h, const float* __restrict__ wl,
                 const float* __restrict__ wr, const float* __restrict__ wlin,
                 float* __restrict__ xlo, float* __restrict__ xro,
                 float* __restrict__ xlino, int N_) {
    int n    = (int)((blockIdx.x * (size_t)blockDim.x + threadIdx.x) >> 5);
    int lane = threadIdx.x & 31;
    if (n >= N_) return;
    float4 u = ((const float4*)(h + (size_t)n * DD))[lane];
    float4 a = ((const float4*)wl)[lane];
    float4 b = ((const float4*)wr)[lane];
    float4 c = ((const float4*)wlin)[lane];
    float s1 = u.x*a.x + u.y*a.y + u.z*a.z + u.w*a.w;
    float s2 = u.x*b.x + u.y*b.y + u.z*b.z + u.w*b.w;
    float s3 = u.x*c.x + u.y*c.y + u.z*c.z + u.w*c.w;
    for (int off = 16; off > 0; off >>= 1) {
        s1 += __shfl_down(s1, off, 32);
        s2 += __shfl_down(s2, off, 32);
        s3 += __shfl_down(s3, off, 32);
    }
    if (lane == 0) { xlo[n] = s1; xro[n] = s2; xlino[n] = s3; }
}

__global__ void edge_out_logits(const float* __restrict__ xlo, const float* __restrict__ xro,
                                const float* __restrict__ atto, const int* __restrict__ src,
                                const int* __restrict__ dst, float* __restrict__ e,
                                float* __restrict__ m, int E_) {
    int j = blockIdx.x * blockDim.x + threadIdx.x;
    if (j >= E_) return;
    float t = xlo[src[j]] + xro[dst[j]];
    t = (t > 0.0f ? t : SLOPE * t) * atto[0];
    e[j] = t;
    atomicMaxF(m + dst[j], t);
}

__global__ void edge_out_acc(const float* __restrict__ xlo, const float* __restrict__ e,
                             const int* __restrict__ src, const int* __restrict__ dst,
                             const float* __restrict__ m, float* __restrict__ s,
                             float* __restrict__ num, int E_) {
    int j = blockIdx.x * blockDim.x + threadIdx.x;
    if (j >= E_) return;
    int d = dst[j];
    float p = __expf(e[j] - m[d]);
    atomicAdd(s + d, p);
    atomicAdd(num + d, p * xlo[src[j]]);
}

__global__ void node_out_final(const float* __restrict__ num, const float* __restrict__ s,
                               const float* __restrict__ bo, const float* __restrict__ xlino,
                               const float* __restrict__ blino, float* __restrict__ out, int N_) {
    int n = blockIdx.x * blockDim.x + threadIdx.x;
    if (n >= N_) return;
    out[n] = num[n] / (s[n] + 1e-16f) + bo[0] + xlino[n] + blino[0];
}

// ---------------------------------------------------------------------------
// Orchestration
// ---------------------------------------------------------------------------
extern "C" void kernel_launch(void* const* d_in, const int* in_sizes, int n_in,
                              void* d_out, int out_size, void* d_ws, size_t ws_size,
                              hipStream_t stream) {
    (void)n_in; (void)out_size; (void)ws_size;
    const float* x   = (const float*)d_in[0];
    const int*   ei  = (const int*)d_in[1];
    const int N_ = in_sizes[0] / DD;
    const int E_ = in_sizes[1] / 2;
    const int* src = ei;
    const int* dst = ei + E_;

    const float *Wl1=(const float*)d_in[2],  *Wr1=(const float*)d_in[3];
    const float *att1=(const float*)d_in[4], *b1=(const float*)d_in[5];
    const float *Wlin1=(const float*)d_in[6],*blin1=(const float*)d_in[7];
    const float *Wl2=(const float*)d_in[8],  *Wr2=(const float*)d_in[9];
    const float *att2=(const float*)d_in[10],*b2=(const float*)d_in[11];
    const float *Wlin2=(const float*)d_in[12],*blin2=(const float*)d_in[13];
    const float *Wlo=(const float*)d_in[14], *Wro=(const float*)d_in[15];
    const float *atto=(const float*)d_in[16],*bo=(const float*)d_in[17];
    const float *Wlino=(const float*)d_in[18],*blino=(const float*)d_in[19];

    const size_t ND = (size_t)N_ * DD;
    float* ws     = (float*)d_ws;
    float* bufXL  = ws;
    float* bufXR  = bufXL  + ND;
    float* bufLIN = bufXR  + ND;
    float* bufNUM = bufLIN + ND;
    float* bufH   = bufNUM + ND;
    float* bm     = bufH   + ND;
    float* bs     = bm + N_;
    float* be     = bs + N_;
    float* xlo    = be + E_;
    float* xro    = xlo + N_;
    float* xlino  = xro + N_;
    float* numo   = xlino + N_;
    // bf16 region (64B-align the boundary)
    size_t fOff   = (size_t)((numo + N_) - ws);
    fOff          = (fOff + 15) & ~(size_t)15;
    bf16* bfBase  = (bf16*)(ws + fOff);
    bf16* Xhi     = bfBase;
    bf16* Xlo     = Xhi + ND;
    bf16* wfrag   = Xlo + ND;          // 6 packed weights, WFRAG_TOTAL each

    const float NEG_INF = -__builtin_huge_valf();
    const int gemmBlocks     = ((N_ + 15) / 16 + 7) / 8;
    const int edgeWaveBlocks = (int)(((size_t)E_ * 32 + 255) / 256);
    const int edgeBlocks     = (E_ + 255) / 256;
    const int ndBlocks       = (int)((ND + 255) / 256);
    const int nBlocks        = (N_ + 255) / 256;
    const int nWaveBlocks    = (int)(((size_t)N_ * 32 + 255) / 256);
    const int packBlocks     = (WFRAG_ELEMS + 255) / 256;

    // Pack all six 128x128 weights into fragment order once.
    const float* bigW[6] = { Wl1, Wr1, Wlin1, Wl2, Wr2, Wlin2 };
    for (int w = 0; w < 6; ++w)
        pack_w_frags<<<packBlocks, 256, 0, stream>>>(bigW[w], wfrag + (size_t)w * WFRAG_TOTAL);

    auto run_gat = [&](const float* in, int wBase, const float* att, const float* b,
                       const float* blin, float* out) {
        split_bf16<<<ndBlocks, 256, 0, stream>>>(in, Xhi, Xlo, (int)ND);
        gemm128_wmma<<<gemmBlocks, 256, 0, stream>>>(Xhi, Xlo,
            wfrag + (size_t)(wBase+0) * WFRAG_TOTAL, nullptr, bufXL,  N_);
        gemm128_wmma<<<gemmBlocks, 256, 0, stream>>>(Xhi, Xlo,
            wfrag + (size_t)(wBase+1) * WFRAG_TOTAL, nullptr, bufXR,  N_);
        gemm128_wmma<<<gemmBlocks, 256, 0, stream>>>(Xhi, Xlo,
            wfrag + (size_t)(wBase+2) * WFRAG_TOTAL, blin,    bufLIN, N_);
        fill_f32<<<nBlocks, 256, 0, stream>>>(bm, NEG_INF, N_);
        fill_f32<<<nBlocks, 256, 0, stream>>>(bs, 0.0f, N_);
        fill_f32<<<ndBlocks, 256, 0, stream>>>(bufNUM, 0.0f, (int)ND);
        edge_logits_max<<<edgeWaveBlocks, 256, 0, stream>>>(bufXL, bufXR, att, src, dst, be, bm, E_);
        edge_softmax_acc<<<edgeWaveBlocks, 256, 0, stream>>>(bufXL, be, src, dst, bm, bs, bufNUM, E_);
        node_combine_relu<<<ndBlocks, 256, 0, stream>>>(bufNUM, bs, b, bufLIN, out, (int)ND);
    };

    // Layer 1: x -> bufH ; Layer 2: bufH -> bufH (combine doesn't read input)
    run_gat(x,    0, att1, b1, blin1, bufH);
    run_gat(bufH, 3, att2, b2, blin2, bufH);

    // Output head (D -> 1)
    dot3_kernel<<<nWaveBlocks, 256, 0, stream>>>(bufH, Wlo, Wro, Wlino, xlo, xro, xlino, N_);
    fill_f32<<<nBlocks, 256, 0, stream>>>(bm, NEG_INF, N_);
    fill_f32<<<nBlocks, 256, 0, stream>>>(bs, 0.0f, N_);
    fill_f32<<<nBlocks, 256, 0, stream>>>(numo, 0.0f, N_);
    edge_out_logits<<<edgeBlocks, 256, 0, stream>>>(xlo, xro, atto, src, dst, be, bm, E_);
    edge_out_acc<<<edgeBlocks, 256, 0, stream>>>(xlo, be, src, dst, bm, bs, numo, E_);
    node_out_final<<<nBlocks, 256, 0, stream>>>(numo, bs, bo, xlino, blino, (float*)d_out, N_);
}
